// GazeGuidedAttention_47914655154437
// MI455X (gfx1250) — matrix operand order repaired and checked by reading, not compiled, not run on hardware
//
#include <hip/hip_runtime.h>
#include <stdint.h>

#define BB 2
#define NN 2048
#define DD 1024
#define HH 16
#define HD 64

typedef __attribute__((ext_vector_type(16))) __bf16 v16bf;
typedef __attribute__((ext_vector_type(8)))  float  v8f;

union ABfrag { uint4 q[2]; v16bf v; };

__device__ __forceinline__ unsigned short f2bf(float f) {
  union { float f; unsigned int u; } c; c.f = f;
  unsigned int u = c.u;
  u += 0x7fffu + ((u >> 16) & 1u);   // round-to-nearest-even
  return (unsigned short)(u >> 16);
}

__device__ __forceinline__ v8f wmma_bf16(const ABfrag& a, const ABfrag& b, v8f c) {
  return __builtin_amdgcn_wmma_f32_16x16x32_bf16(false, a.v, false, b.v,
                                                 (short)0, c, false, false);
}

// ---------------------------------------------------------------- kernel 0
__global__ void convert_weights(const float* __restrict__ wq,
                                const float* __restrict__ wp,
                                unsigned short* __restrict__ oq,
                                unsigned short* __restrict__ op) {
  int i = blockIdx.x * 256 + threadIdx.x;
  if (i < DD * 3 * DD) oq[i] = f2bf(wq[i]);
  if (i < DD * DD)     op[i] = f2bf(wp[i]);
}

// ---------------------------------------------------------------- kernel 1
// qkv = x @ Wqkv + b ; scatter into Q[b][h][n][d], K[b][h][n][d], Vt[b][h][d][n] (bf16)
__global__ __launch_bounds__(128) void qkv_gemm(
    const float* __restrict__ x, const unsigned short* __restrict__ Wb,
    const float* __restrict__ bq,
    unsigned short* __restrict__ Qb, unsigned short* __restrict__ Kb,
    unsigned short* __restrict__ Vtb) {
  __shared__ alignas(16) unsigned short As[64 * 40];
  __shared__ alignas(16) unsigned short Bs[64 * 40];
  const int NOUT = 3 * DD;
  int lane = threadIdx.x & 31, wave = threadIdx.x >> 5;
  int hi = lane >> 4, ln = lane & 15;
  int wm = wave >> 1, wn = wave & 1;
  int row0 = blockIdx.y * 64, col0 = blockIdx.x * 64;

  v8f acc[2][2] = {};
  for (int k0 = 0; k0 < DD; k0 += 32) {
#pragma unroll
    for (int it = 0; it < 4; ++it) {           // A tile: 64x32 f32 -> bf16
      int flat = it * 128 + threadIdx.x;
      int r = flat >> 3, c4 = (flat & 7) << 2;
      float4 f = *(const float4*)&x[(size_t)(row0 + r) * DD + k0 + c4];
      int o = r * 40 + c4;
      As[o] = f2bf(f.x); As[o + 1] = f2bf(f.y);
      As[o + 2] = f2bf(f.z); As[o + 3] = f2bf(f.w);
    }
#pragma unroll
    for (int it = 0; it < 4; ++it) {           // B tile: 32x64, stored [col][k]
      int flat = it * 128 + threadIdx.x;
      int kr = flat >> 4, g = (flat & 15) << 2;
      uint2 w2 = *(const uint2*)&Wb[(size_t)(k0 + kr) * NOUT + col0 + g];
      Bs[(g + 0) * 40 + kr] = (unsigned short)(w2.x & 0xffff);
      Bs[(g + 1) * 40 + kr] = (unsigned short)(w2.x >> 16);
      Bs[(g + 2) * 40 + kr] = (unsigned short)(w2.y & 0xffff);
      Bs[(g + 3) * 40 + kr] = (unsigned short)(w2.y >> 16);
    }
    __syncthreads();
    ABfrag a[2], b[2];
#pragma unroll
    for (int fm = 0; fm < 2; ++fm) {           // A frag: m=lane&15, k per ISA layout
      int base = (wm * 32 + fm * 16 + ln) * 40;
      a[fm].q[0] = *(const uint4*)&As[base + 8 * hi];
      a[fm].q[1] = *(const uint4*)&As[base + 16 + 8 * hi];
    }
#pragma unroll
    for (int fn = 0; fn < 2; ++fn) {           // B frag: n=lane&15, k per ISA layout
      int base = (wn * 32 + fn * 16 + ln) * 40 + 8 * hi;
      b[fn].q[0] = *(const uint4*)&Bs[base];
      b[fn].q[1] = *(const uint4*)&Bs[base + 16];
    }
#pragma unroll
    for (int fm = 0; fm < 2; ++fm)
#pragma unroll
      for (int fn = 0; fn < 2; ++fn)
        acc[fm][fn] = wmma_bf16(a[fm], b[fn], acc[fm][fn]);
    __syncthreads();
  }
  // epilogue: sel/h/b are uniform per (block, fn) -> hoist pointer selection
  int b_ = row0 >> 11;
  int nbase = row0 & 2047;
#pragma unroll
  for (int fn = 0; fn < 2; ++fn) {
    int j = col0 + wn * 32 + fn * 16 + ln;
    float bj = bq[j];
    int sel = j >> 10, within = j & 1023;
    int h = within >> 6, d = within & 63;
    int bh = b_ * HH + h;
    unsigned short* dst;
    size_t strideN;
    if (sel == 0)      { dst = Qb  + (size_t)bh * NN * HD + d; strideN = HD; }
    else if (sel == 1) { dst = Kb  + (size_t)bh * NN * HD + d; strideN = HD; }
    else               { dst = Vtb + ((size_t)bh * HD + d) * NN; strideN = 1; }
#pragma unroll
    for (int fm = 0; fm < 2; ++fm)
#pragma unroll
      for (int r = 0; r < 8; ++r) {
        int n_ = nbase + wm * 32 + fm * 16 + r + 8 * hi;
        dst[(size_t)n_ * strideN] = f2bf(acc[fm][fn][r] + bj);
      }
  }
}

// ---------------------------------------------------------------- kernel 2
// Flash attention: one block = 4 waves = 64 queries of one (b,h).
__global__ __launch_bounds__(128) void attn_kernel(
    const unsigned short* __restrict__ Qb, const unsigned short* __restrict__ Kb,
    const unsigned short* __restrict__ Vtb, const float* __restrict__ abias,
    unsigned short* __restrict__ Ao) {
  __shared__ alignas(16) unsigned short Plds[4 * 16 * 40];
  int lane = threadIdx.x & 31, wave = threadIdx.x >> 5;
  int hi = lane >> 4, ln = lane & 15;
  int qt = blockIdx.x & 31;            // N/64 = 32 query tiles
  int bh = blockIdx.x >> 5;            // 0..31
  int b = bh >> 4;
  int h = bh & 15;
  int q0 = qt * 64 + wave * 16;
  const unsigned short* Qp = Qb + (size_t)bh * NN * HD;
  const unsigned short* Kp = Kb + (size_t)bh * NN * HD;
  const unsigned short* Vp = Vtb + (size_t)bh * HD * NN;
  unsigned short* P = &Plds[wave * 16 * 40];

  ABfrag aq0, aq1;                     // Q tile 16x64 as two A fragments
  {
    const unsigned short* qrow = Qp + (size_t)(q0 + ln) * HD;
    aq0.q[0] = *(const uint4*)&qrow[8 * hi];
    aq0.q[1] = *(const uint4*)&qrow[16 + 8 * hi];
    aq1.q[0] = *(const uint4*)&qrow[32 + 8 * hi];
    aq1.q[1] = *(const uint4*)&qrow[48 + 8 * hi];
  }

  v8f O[4] = {};                       // 16x64 output accumulator
  float mrow[8], lrow[8];
#pragma unroll
  for (int r = 0; r < 8; ++r) { mrow[r] = -1e30f; lrow[r] = 0.f; }

  for (int kb = 0; kb < NN; kb += 32) {
    if (kb + 32 < NN) {                // warm next K tile (global_prefetch_b8)
      __builtin_prefetch(Kp + (size_t)(kb + 32 + ln) * HD, 0, 1);
      __builtin_prefetch(Kp + (size_t)(kb + 48 + ln) * HD, 0, 1);
    }
    ABfrag bk[2][2];                   // [key-16-tile][d-half]
#pragma unroll
    for (int t = 0; t < 2; ++t) {
      const unsigned short* krow = Kp + (size_t)(kb + t * 16 + ln) * HD;
#pragma unroll
      for (int dh = 0; dh < 2; ++dh) {
        bk[t][dh].q[0] = *(const uint4*)&krow[dh * 32 + 8 * hi];
        bk[t][dh].q[1] = *(const uint4*)&krow[dh * 32 + 16 + 8 * hi];
      }
    }
    v8f S0 = {}, S1 = {};
    S0 = wmma_bf16(aq0, bk[0][0], S0);
    S0 = wmma_bf16(aq1, bk[0][1], S0);
    S1 = wmma_bf16(aq0, bk[1][0], S1);
    S1 = wmma_bf16(aq1, bk[1][1], S1);

    float bias0 = abias[b * NN + kb + ln];
    float bias1 = abias[b * NN + kb + 16 + ln];

    float alpha[8];
#pragma unroll
    for (int r = 0; r < 8; ++r) {      // online softmax, rows r+8*hi
      float s0 = S0[r] * 0.125f + bias0;
      float s1 = S1[r] * 0.125f + bias1;
      float mx = fmaxf(s0, s1);
#pragma unroll
      for (int msk = 8; msk >= 1; msk >>= 1)
        mx = fmaxf(mx, __shfl_xor(mx, msk, 32));   // half-wave row reduce
      float mnew = fmaxf(mrow[r], mx);
      float a = __expf(mrow[r] - mnew);
      float p0 = __expf(s0 - mnew), p1 = __expf(s1 - mnew);
      float rs = p0 + p1;
#pragma unroll
      for (int msk = 8; msk >= 1; msk >>= 1)
        rs += __shfl_xor(rs, msk, 32);
      lrow[r] = lrow[r] * a + rs;
      mrow[r] = mnew;
      alpha[r] = a;
      int m = r + 8 * hi;              // P tile 16x32 bf16 in per-wave LDS
      P[m * 40 + ln] = f2bf(p0);
      P[m * 40 + 16 + ln] = f2bf(p1);
    }
#pragma unroll
    for (int fd = 0; fd < 4; ++fd)
#pragma unroll
      for (int r = 0; r < 8; ++r) O[fd][r] *= alpha[r];

    ABfrag ap;                         // P as A fragment (same-wave DS is in-order)
    {
      int base = ln * 40;
      ap.q[0] = *(const uint4*)&P[base + 8 * hi];
      ap.q[1] = *(const uint4*)&P[base + 16 + 8 * hi];
    }
#pragma unroll
    for (int fd = 0; fd < 4; ++fd) {   // O += P @ V, V transposed -> contiguous
      const unsigned short* vrow =
          Vp + (size_t)(fd * 16 + ln) * NN + kb + 8 * hi;
      ABfrag bv;
      bv.q[0] = *(const uint4*)&vrow[0];
      bv.q[1] = *(const uint4*)&vrow[16];
      O[fd] = wmma_bf16(ap, bv, O[fd]);
    }
  }
#pragma unroll
  for (int fd = 0; fd < 4; ++fd)
#pragma unroll
    for (int r = 0; r < 8; ++r) {
      int q = q0 + r + 8 * hi;
      int d = fd * 16 + ln;
      float v = O[fd][r] / lrow[r];
      Ao[((size_t)(b * NN + q)) * DD + h * HD + d] = f2bf(v);
    }
}

// ---------------------------------------------------------------- kernel 3
// out = Aout @ Wproj + b_proj (f32 output)
// A tile is bf16 row-major in both global and LDS -> use async copy to LDS.
__global__ __launch_bounds__(128) void proj_gemm(
    const unsigned short* __restrict__ Ab, const unsigned short* __restrict__ Wb,
    const float* __restrict__ bp, float* __restrict__ out) {
  __shared__ alignas(16) unsigned short As[64 * 40];
  __shared__ alignas(16) unsigned short Bs[64 * 40];
  const int NOUT = DD;
  int lane = threadIdx.x & 31, wave = threadIdx.x >> 5;
  int hi = lane >> 4, ln = lane & 15;
  int wm = wave >> 1, wn = wave & 1;
  int row0 = blockIdx.y * 64, col0 = blockIdx.x * 64;

  v8f acc[2][2] = {};
  for (int k0 = 0; k0 < DD; k0 += 32) {
#pragma unroll
    for (int it = 0; it < 2; ++it) {           // A tile 64x32 bf16, async -> LDS
      int flat = it * 128 + threadIdx.x;       // 256 chunks of 16B
      int r = flat >> 2, c = (flat & 3) << 3;
      unsigned long long g =
          (unsigned long long)(uintptr_t)&Ab[(size_t)(row0 + r) * DD + k0 + c];
      unsigned lo = (unsigned)(uintptr_t)&As[r * 40 + c];
      asm volatile("global_load_async_to_lds_b128 %0, %1, off"
                   :: "v"(lo), "v"(g) : "memory");
    }
#pragma unroll
    for (int it = 0; it < 4; ++it) {           // B tile 32x64 -> [col][k]
      int flat = it * 128 + threadIdx.x;
      int kr = flat >> 4, g = (flat & 15) << 2;
      uint2 w2 = *(const uint2*)&Wb[(size_t)(k0 + kr) * NOUT + col0 + g];
      Bs[(g + 0) * 40 + kr] = (unsigned short)(w2.x & 0xffff);
      Bs[(g + 1) * 40 + kr] = (unsigned short)(w2.x >> 16);
      Bs[(g + 2) * 40 + kr] = (unsigned short)(w2.y & 0xffff);
      Bs[(g + 3) * 40 + kr] = (unsigned short)(w2.y >> 16);
    }
    asm volatile("s_wait_asynccnt 0x0" ::: "memory");
    __syncthreads();
    ABfrag a[2], b[2];
#pragma unroll
    for (int fm = 0; fm < 2; ++fm) {
      int base = (wm * 32 + fm * 16 + ln) * 40;
      a[fm].q[0] = *(const uint4*)&As[base + 8 * hi];
      a[fm].q[1] = *(const uint4*)&As[base + 16 + 8 * hi];
    }
#pragma unroll
    for (int fn = 0; fn < 2; ++fn) {
      int base = (wn * 32 + fn * 16 + ln) * 40 + 8 * hi;
      b[fn].q[0] = *(const uint4*)&Bs[base];
      b[fn].q[1] = *(const uint4*)&Bs[base + 16];
    }
#pragma unroll
    for (int fm = 0; fm < 2; ++fm)
#pragma unroll
      for (int fn = 0; fn < 2; ++fn)
        acc[fm][fn] = wmma_bf16(a[fm], b[fn], acc[fm][fn]);
    __syncthreads();
  }
#pragma unroll
  for (int fm = 0; fm < 2; ++fm)
#pragma unroll
    for (int fn = 0; fn < 2; ++fn) {
      int j = col0 + wn * 32 + fn * 16 + ln;
      float bj = bp[j];
#pragma unroll
      for (int r = 0; r < 8; ++r) {
        int row = row0 + wm * 32 + fm * 16 + r + 8 * hi;
        out[(size_t)row * DD + j] = acc[fm][fn][r] + bj;
      }
    }
}

// ---------------------------------------------------------------- launcher
extern "C" void kernel_launch(void* const* d_in, const int* in_sizes, int n_in,
                              void* d_out, int out_size, void* d_ws, size_t ws_size,
                              hipStream_t stream) {
  const float* x  = (const float*)d_in[0];
  const float* ab = (const float*)d_in[1];
  const float* wq = (const float*)d_in[2];
  const float* bq = (const float*)d_in[3];
  const float* wp = (const float*)d_in[4];
  const float* bp = (const float*)d_in[5];
  float* out = (float*)d_out;
  char* ws = (char*)d_ws;

  unsigned short* Wqkv  = (unsigned short*)(ws);               // 6,291,456 B
  unsigned short* Wproj = (unsigned short*)(ws + 6291456);     // 2,097,152 B
  unsigned short* Qb    = (unsigned short*)(ws + 8388608);     // 8,388,608 B
  unsigned short* Kb    = (unsigned short*)(ws + 16777216);    // 8,388,608 B
  unsigned short* Vtb   = (unsigned short*)(ws + 25165824);    // 8,388,608 B
  unsigned short* Ao    = (unsigned short*)(ws + 33554432);    // 8,388,608 B

  convert_weights<<<(DD * 3 * DD + 255) / 256, 256, 0, stream>>>(wq, wp, Wqkv, Wproj);
  qkv_gemm<<<dim3(3 * DD / 64, BB * NN / 64), 128, 0, stream>>>(x, Wqkv, bq, Qb, Kb, Vtb);
  attn_kernel<<<BB * HH * (NN / 64), 128, 0, stream>>>(Qb, Kb, Vtb, ab, Ao);
  proj_gemm<<<dim3(DD / 64, BB * NN / 64), 128, 0, stream>>>(Ao, Wproj, bp, out);
}
